// Attention_81922206204606
// MI455X (gfx1250) — matrix operand order repaired and checked by reading, not compiled, not run on hardware
//
#include <hip/hip_runtime.h>

typedef __bf16 bf16;
typedef __attribute__((ext_vector_type(16))) __bf16 v16bf;
typedef __attribute__((ext_vector_type(8)))  __bf16 v8bf;
typedef __attribute__((ext_vector_type(8)))  float  v8f;

#define N_SEQ 2048
#define NH    16
#define DH    64
#define DMODEL 1024

// ---------------------------------------------------------------------------
// WMMA helpers (bf16 16x16x32, f32 accumulate)
// ---------------------------------------------------------------------------
__device__ __forceinline__ v8f wmma_bf16(v16bf a, v16bf b, v8f c) {
  return __builtin_amdgcn_wmma_f32_16x16x32_bf16(
      /*neg_a=*/false, a, /*neg_b=*/false, b,
      /*c_mod=*/(short)0, c, /*reuse_a=*/false, /*reuse_b=*/false);
}

// Load a 16x32 bf16 A-operand (lanes 0-15 M=lane hold K chunks
// {k0..k0+15}; lanes 16-31 M=lane-16 hold {k0+16..k0+31}).
// `base` is row-major with `rowStride` elements per row; `row` is the lane's M row.
__device__ __forceinline__ v16bf load_a16(const bf16* base, int rowStride,
                                          int row, int k0, int lane) {
  const bf16* r = base + (size_t)row * rowStride;
  int koff = k0 + ((lane & 16) ? 16 : 0);
  union { v16bf v; v8bf h[2]; } u;
  u.h[0] = *(const v8bf*)(r + koff);
  u.h[1] = *(const v8bf*)(r + koff + 8);
  return u.v;
}

// Intra-16-lane-group reductions / scans (wave32; groups {0..15},{16..31})
__device__ __forceinline__ float grp16_max(float x) {
#pragma unroll
  for (int m = 1; m < 16; m <<= 1) x = fmaxf(x, __shfl_xor(x, m, 32));
  return x;
}
__device__ __forceinline__ float grp16_sum(float x) {
#pragma unroll
  for (int m = 1; m < 16; m <<= 1) x += __shfl_xor(x, m, 32);
  return x;
}
// Inclusive suffix sum across the 16 lanes of a group (lane%16 == column)
__device__ __forceinline__ float grp16_suffix(float x, int lane) {
#pragma unroll
  for (int d = 1; d < 16; d <<= 1) {
    float t = __shfl_down(x, d, 16);
    if ((lane & 15) + d < 16) x += t;
  }
  return x;
}

// ---------------------------------------------------------------------------
// Prep kernels: f32 -> bf16 convert / transpose
// ---------------------------------------------------------------------------
__global__ void cvt_bf16_kernel(const float* __restrict__ in,
                                bf16* __restrict__ out, int n) {
  int i = blockIdx.x * 256 + threadIdx.x;
  if (i < n) out[i] = (bf16)in[i];
}

// in: R x C row-major f32 ; out: C x R row-major bf16 (out[c][r] = in[r][c])
__global__ void transpose_bf16_kernel(const float* __restrict__ in,
                                      bf16* __restrict__ out, int R, int C) {
  int i = blockIdx.x * 256 + threadIdx.x;
  if (i < R * C) {
    int r = i / C, c = i % C;
    out[(size_t)c * R + r] = (bf16)in[i];
  }
}

// ---------------------------------------------------------------------------
// Fused QKV projection: C = Xbf(2048x1024) @ W ; W = WqT / WkvT (col-major).
// Each wave computes a 16x64 tile == one head's 64 dims for Q, K, or V.
// Q,K stored row-major per head; V stored transposed per head (Vt[h][d][j]).
// ---------------------------------------------------------------------------
__global__ __launch_bounds__(128) void qkv_kernel(
    const bf16* __restrict__ Xbf, const bf16* __restrict__ WqT,
    const bf16* __restrict__ WkvT, bf16* __restrict__ Qbf,
    bf16* __restrict__ Kbf, bf16* __restrict__ Vt) {
  const int lane = threadIdx.x & 31;
  const int wid  = threadIdx.x >> 5;
  const int wg   = blockIdx.x * 4 + wid;   // 128 row-blocks * 48 col-blocks
  const int cb   = wg % 48;
  const int i0   = (wg / 48) * 16;
  const int col  = lane & 15;
  const int khalf = (lane & 16) ? 16 : 0;

  const bf16* WT;
  int wcol;
  if (cb < 16)      { WT = WqT;  wcol = cb * 64; }
  else if (cb < 32) { WT = WkvT; wcol = (cb - 16) * 64; }
  else              { WT = WkvT; wcol = 1024 + (cb - 32) * 64; }

  v8f acc[4] = {v8f{}, v8f{}, v8f{}, v8f{}};
  for (int k0 = 0; k0 < DMODEL; k0 += 32) {
    v16bf a = load_a16(Xbf, DMODEL, i0 + (lane & 15), k0, lane);
#pragma unroll
    for (int t = 0; t < 4; ++t) {
      const bf16* wrow = WT + (size_t)(wcol + t * 16 + col) * DMODEL + k0;
      v16bf b = *(const v16bf*)(wrow + khalf);
      acc[t] = wmma_bf16(a, b, acc[t]);
    }
  }

  const int h = cb % 16;
#pragma unroll
  for (int t = 0; t < 4; ++t) {
#pragma unroll
    for (int v = 0; v < 8; ++v) {
      int r = v + ((lane & 16) ? 8 : 0);
      int d = t * 16 + col;
      bf16 val = (bf16)acc[t][v];
      if (cb < 16)
        Qbf[(size_t)h * N_SEQ * DH + (size_t)(i0 + r) * DH + d] = val;
      else if (cb < 32)
        Kbf[(size_t)h * N_SEQ * DH + (size_t)(i0 + r) * DH + d] = val;
      else
        Vt[(size_t)h * DH * N_SEQ + (size_t)d * N_SEQ + (i0 + r)] = val;
    }
  }
}

// ---------------------------------------------------------------------------
// Flash-style CoPE attention. One wave per 16-row query block. Keys iterated
// HIGH -> LOW j in steps of 32 so the CoPE suffix cumsum can be carried.
// ---------------------------------------------------------------------------
__global__ __launch_bounds__(128) void attn_kernel(
    const bf16* __restrict__ Qbf, const bf16* __restrict__ Kbf,
    const bf16* __restrict__ Vt, const bf16* __restrict__ posT,
    bf16* __restrict__ AttnOut) {
  __shared__ float e_s[4][16][64];   // E = Q @ pos_emb tile, per wave
  __shared__ bf16  p_s[4][16][32];   // attn-prob repack buffer, per wave

  const int lane = threadIdx.x & 31;
  const int wid  = threadIdx.x >> 5;
  const int h    = blockIdx.y;
  const int i0   = (blockIdx.x * 4 + wid) * 16;
  const int col  = lane & 15;
  const int khalf = (lane & 16) ? 16 : 0;
  const float scale = 0.125f;  // 64^-0.5

  const bf16* Qh  = Qbf + (size_t)h * N_SEQ * DH;
  const bf16* Kh  = Kbf + (size_t)h * N_SEQ * DH;
  const bf16* Vth = Vt  + (size_t)h * DH * N_SEQ;

  // Q block as A operands for K-chunks [0,32) and [32,64)
  v16bf qa0 = load_a16(Qh, DH, i0 + (lane & 15), 0, lane);
  v16bf qa1 = load_a16(Qh, DH, i0 + (lane & 15), 32, lane);

  // E(16x64) = Qblk(16x64) @ pos_emb(64x64), via WMMA into LDS (f32)
#pragma unroll
  for (int t = 0; t < 4; ++t) {
    const bf16* prow = posT + (size_t)(t * 16 + col) * DH;  // posT[t][c]
    v16bf b0 = *(const v16bf*)(prow + khalf);
    v16bf b1 = *(const v16bf*)(prow + 32 + khalf);
    v8f e = {};
    e = wmma_bf16(qa0, b0, e);
    e = wmma_bf16(qa1, b1, e);
#pragma unroll
    for (int v = 0; v < 8; ++v) {
      int r = v + ((lane & 16) ? 8 : 0);
      e_s[wid][r][t * 16 + col] = e[v];
    }
  }

  float carry[8], mrow[8], lrow[8];
  v8f oacc[4] = {v8f{}, v8f{}, v8f{}, v8f{}};
#pragma unroll
  for (int v = 0; v < 8; ++v) { carry[v] = 0.f; mrow[v] = -1e30f; lrow[v] = 0.f; }

  for (int j0 = N_SEQ - 32; j0 >= 0; j0 -= 32) {
    float st[2][8];
    // sub-tile 1 (keys j0+16..j0+31) first, then sub-tile 0 — suffix order
#pragma unroll
    for (int s = 1; s >= 0; --s) {
      const bf16* krow = Kh + (size_t)(j0 + s * 16 + col) * DH;
      v16bf b0 = *(const v16bf*)(krow + khalf);
      v16bf b1 = *(const v16bf*)(krow + 32 + khalf);
      v8f c = {};
      c = wmma_bf16(qa0, b0, c);
      c = wmma_bf16(qa1, b1, c);
#pragma unroll
      for (int v = 0; v < 8; ++v) {
        float sim = c[v] * scale;
        float g   = 1.0f / (1.0f + __expf(-sim));      // sigmoid gate
        float suf = grp16_suffix(g, lane);             // in-tile suffix sum
        float P   = carry[v] + suf;
        carry[v] += __shfl(suf, lane & 16, 32);        // col-0 of the group
        P = fminf(P, 63.0f);
        float pf = floorf(P);
        int   ifl = (int)pf;
        int   icl = (int)ceilf(P);
        float w   = P - pf;
        int   r   = v + ((lane & 16) ? 8 : 0);
        float ef  = e_s[wid][r][ifl];                  // LDS gathers
        float ec  = e_s[wid][r][icl];
        st[s][v] = sim + w * ec + (1.0f - w) * ef;
      }
    }

    // Online softmax across these 32 keys
    float p[2][8], alpha[8];
#pragma unroll
    for (int v = 0; v < 8; ++v) {
      float tmax = grp16_max(fmaxf(st[0][v], st[1][v]));
      float mnew = fmaxf(mrow[v], tmax);
      float a    = __expf(mrow[v] - mnew);
      p[0][v] = __expf(st[0][v] - mnew);
      p[1][v] = __expf(st[1][v] - mnew);
      float rs = grp16_sum(p[0][v] + p[1][v]);
      lrow[v] = lrow[v] * a + rs;
      mrow[v] = mnew;
      alpha[v] = a;
    }
#pragma unroll
    for (int t = 0; t < 4; ++t)
#pragma unroll
      for (int v = 0; v < 8; ++v) oacc[t][v] *= alpha[v];

    // Repack probs (C layout -> A layout) through LDS
#pragma unroll
    for (int s = 0; s < 2; ++s)
#pragma unroll
      for (int v = 0; v < 8; ++v) {
        int r = v + ((lane & 16) ? 8 : 0);
        p_s[wid][r][s * 16 + col] = (bf16)p[s][v];
      }
    v16bf pa = load_a16(&p_s[wid][0][0], 32, lane & 15, 0, lane);

    // O(16x64) += P(16x32) @ V(32x64); B columns contiguous via Vt
#pragma unroll
    for (int t = 0; t < 4; ++t) {
      const bf16* vrow = Vth + (size_t)(t * 16 + col) * N_SEQ + j0;
      v16bf bv = *(const v16bf*)(vrow + khalf);
      oacc[t] = wmma_bf16(pa, bv, oacc[t]);
    }
  }

  // Normalize and write merged-head bf16 output row-major (for out-proj GEMM)
#pragma unroll
  for (int t = 0; t < 4; ++t)
#pragma unroll
    for (int v = 0; v < 8; ++v) {
      int r = v + ((lane & 16) ? 8 : 0);
      float o = oacc[t][v] / lrow[v];
      AttnOut[(size_t)(i0 + r) * DMODEL + h * DH + t * 16 + col] = (bf16)o;
    }
}

// ---------------------------------------------------------------------------
// Output projection: Out = AttnOut(2048x1024) @ Wout + b_out (f32 result)
// ---------------------------------------------------------------------------
__global__ __launch_bounds__(128) void outproj_kernel(
    const bf16* __restrict__ AttnOut, const bf16* __restrict__ WoutT,
    const float* __restrict__ b_out, float* __restrict__ Out) {
  const int lane = threadIdx.x & 31;
  const int wid  = threadIdx.x >> 5;
  const int wg   = blockIdx.x * 4 + wid;    // 128 row-blocks * 16 col-blocks
  const int cb   = wg % 16;
  const int i0   = (wg / 16) * 16;
  const int n0   = cb * 64;
  const int col  = lane & 15;
  const int khalf = (lane & 16) ? 16 : 0;

  v8f acc[4] = {v8f{}, v8f{}, v8f{}, v8f{}};
  for (int k0 = 0; k0 < DMODEL; k0 += 32) {
    v16bf a = load_a16(AttnOut, DMODEL, i0 + (lane & 15), k0, lane);
#pragma unroll
    for (int t = 0; t < 4; ++t) {
      const bf16* wrow = WoutT + (size_t)(n0 + t * 16 + col) * DMODEL + k0;
      v16bf b = *(const v16bf*)(wrow + khalf);
      acc[t] = wmma_bf16(a, b, acc[t]);
    }
  }
#pragma unroll
  for (int t = 0; t < 4; ++t) {
    float bias = b_out[n0 + t * 16 + col];
#pragma unroll
    for (int v = 0; v < 8; ++v) {
      int r = v + ((lane & 16) ? 8 : 0);
      Out[(size_t)(i0 + r) * DMODEL + n0 + t * 16 + col] = acc[t][v] + bias;
    }
  }
}

// ---------------------------------------------------------------------------
// Host launcher
// ---------------------------------------------------------------------------
extern "C" void kernel_launch(void* const* d_in, const int* in_sizes, int n_in,
                              void* d_out, int out_size, void* d_ws, size_t ws_size,
                              hipStream_t stream) {
  const float* x     = (const float*)d_in[0];  // (1,2048,1024)
  const float* Wq    = (const float*)d_in[1];  // (1024,1024)
  const float* Wkv   = (const float*)d_in[2];  // (1024,2048)
  const float* Wout  = (const float*)d_in[3];  // (1024,1024)
  const float* bout  = (const float*)d_in[4];  // (1024,)
  const float* pos   = (const float*)d_in[5];  // (64,64)
  float* out = (float*)d_out;

  // Carve scratch
  char*  ws  = (char*)d_ws;
  size_t off = 0;
  auto carve = [&](size_t bytes) { char* p = ws + off; off += (bytes + 255) & ~(size_t)255; return p; };
  bf16* Xbf     = (bf16*)carve((size_t)N_SEQ * DMODEL * 2);
  bf16* WqT     = (bf16*)carve((size_t)DMODEL * DMODEL * 2);
  bf16* WkvT    = (bf16*)carve((size_t)2 * DMODEL * DMODEL * 2);
  bf16* WoutT   = (bf16*)carve((size_t)DMODEL * DMODEL * 2);
  bf16* posT    = (bf16*)carve((size_t)DH * DH * 2);
  bf16* Qbf     = (bf16*)carve((size_t)NH * N_SEQ * DH * 2);
  bf16* Kbf     = (bf16*)carve((size_t)NH * N_SEQ * DH * 2);
  bf16* Vt      = (bf16*)carve((size_t)NH * DH * N_SEQ * 2);
  bf16* AttnOut = (bf16*)carve((size_t)N_SEQ * DMODEL * 2);
  (void)ws_size; (void)n_in; (void)in_sizes; (void)out_size;

  // 1) precision convert + weight transposes
  {
    int n = N_SEQ * DMODEL;
    cvt_bf16_kernel<<<(n + 255) / 256, 256, 0, stream>>>(x, Xbf, n);
  }
  transpose_bf16_kernel<<<(DMODEL * DMODEL + 255) / 256, 256, 0, stream>>>(Wq, WqT, DMODEL, DMODEL);
  transpose_bf16_kernel<<<(DMODEL * 2 * DMODEL + 255) / 256, 256, 0, stream>>>(Wkv, WkvT, DMODEL, 2 * DMODEL);
  transpose_bf16_kernel<<<(DMODEL * DMODEL + 255) / 256, 256, 0, stream>>>(Wout, WoutT, DMODEL, DMODEL);
  transpose_bf16_kernel<<<(DH * DH + 255) / 256, 256, 0, stream>>>(pos, posT, DH, DH);

  // 2) fused QKV projection (6144 waves / 1536 blocks of 4 waves)
  qkv_kernel<<<1536, 128, 0, stream>>>(Xbf, WqT, WkvT, Qbf, Kbf, Vt);

  // 3) flash CoPE attention: 128 query blocks x 16 heads, 4 waves/block
  attn_kernel<<<dim3(32, NH), 128, 0, stream>>>(Qbf, Kbf, Vt, posT, AttnOut);

  // 4) output projection + bias (2048 waves / 512 blocks)
  outproj_kernel<<<512, 128, 0, stream>>>(AttnOut, WoutT, bout, out);
}
